// XLNetRelativeAttentionTP_71957882077386
// MI455X (gfx1250) — matrix operand order repaired and checked by reading, not compiled, not run on hardware
//
#include <hip/hip_runtime.h>

#define QLEN 1024
#define BSZ  2
#define DM   1024
#define NH   16
#define DH   64
#define RLEN 2048

typedef __bf16 bf16_t;
typedef bf16_t v16bf __attribute__((ext_vector_type(16)));
typedef float  v8f   __attribute__((ext_vector_type(8)));

union C8  { v8f v; float f[8]; };
union U16 { v16bf v; uint4 q[2]; unsigned short u[16]; };

__device__ __forceinline__ unsigned short f2bf(float f) {
  unsigned u = __builtin_bit_cast(unsigned, f);
  u += 0x7FFFu + ((u >> 16) & 1u);               // round-to-nearest-even
  return (unsigned short)(u >> 16);
}
__device__ __forceinline__ float bf2f(unsigned short h) {
  unsigned u = ((unsigned)h) << 16;
  return __builtin_bit_cast(float, u);
}
// B-operand tile: lane holds column (lane&15), 16 contiguous K values (K-half by lane>>4)
__device__ __forceinline__ v16bf load16(const unsigned short* p) {
  U16 t; t.q[0] = *(const uint4*)p; t.q[1] = *(const uint4*)(p + 8);
  return t.v;
}
// A-operand tile: lane holds row (lane&15); 16 contiguous K values from k0 + (k0&8), k0 = ks*32 + (lane>=16?8:0)
__device__ __forceinline__ v16bf loadA(const unsigned short* row, int k0) {
  int kk = k0 + (k0 & 8);
  U16 t; t.q[0] = *(const uint4*)(row + kk); t.q[1] = *(const uint4*)(row + kk + 8);
  return t.v;
}
__device__ __forceinline__ v8f wmma_bf(v16bf a, v16bf b, v8f c) {
  return __builtin_amdgcn_wmma_f32_16x16x32_bf16(false, a, false, b, (short)0, c, false, false);
}

// ---------------- prep kernels ----------------

__global__ __launch_bounds__(256) void convert_bf16(const float* __restrict__ src,
                                                    unsigned short* __restrict__ dst, int n) {
  int i = blockIdx.x * 256 + threadIdx.x;
  if (i < n) dst[i] = f2bf(src[i]);
}

// W[k][c] (1024x1024 f32) -> Wt[c][k] (bf16), LDS-tiled transpose
__global__ __launch_bounds__(256) void transpose_conv(const float* __restrict__ W,
                                                      unsigned short* __restrict__ Wt) {
  __shared__ float tile[32][33];
  int tx = threadIdx.x & 31, ty = threadIdx.x >> 5;
  int cb = blockIdx.x * 32, kb = blockIdx.y * 32;
  for (int r = ty; r < 32; r += 8) tile[r][tx] = W[(size_t)(kb + r) * DM + cb + tx];
  __syncthreads();
  for (int r = ty; r < 32; r += 8) Wt[(size_t)(cb + r) * DM + kb + tx] = f2bf(tile[tx][r]);
}

// pack seg_mat[:,:,b,1] into bit words: bits[(b*QLEN+i)*32 + (j>>5)]
__global__ __launch_bounds__(256) void segbits_kernel(const float* __restrict__ seg_mat,
                                                      unsigned* __restrict__ bits) {
  int gid = blockIdx.x * 256 + threadIdx.x;           // over BSZ*QLEN*32
  if (gid >= BSZ * QLEN * 32) return;
  int word = gid & 31, i = (gid >> 5) & (QLEN - 1), b = gid >> 15;
  unsigned acc = 0;
  for (int jj = 0; jj < 32; ++jj) {
    int j = word * 32 + jj;
    float d = seg_mat[(((size_t)i * QLEN + j) * BSZ + b) * 2 + 1];
    acc |= (d > 0.5f ? 1u : 0u) << jj;
  }
  bits[((size_t)b * QLEN + i) * 32 + word] = acc;
}

// ---------------- generic WMMA GEMM (64xK x Kx64 tiles, 4 waves/block) ----------------
// modes: 0=Q (writes qw,qr + ef), 1=K, 2=V(transposed), 3=R, 4=output-proj(+residual, f32)
__global__ __launch_bounds__(128) void wmma_gemm_kernel(
    const unsigned short* __restrict__ A,   // [M][DM] bf16
    const unsigned short* __restrict__ Bt,  // [colGlobal][DM] bf16
    int mode,
    unsigned short* __restrict__ o0, unsigned short* __restrict__ o1,
    float* __restrict__ ef,
    const float* __restrict__ b0, const float* __restrict__ b1,
    const float* __restrict__ b2, const float* __restrict__ se,
    const float* __restrict__ hres, float* __restrict__ resout)
{
  const int w = threadIdx.x >> 5, lane = threadIdx.x & 31;
  const int c = lane & 15, hi = lane >> 4;
  const int colbase = blockIdx.x * 64;
  const int row0 = blockIdx.y * 64 + w * 16;
  const unsigned short* Arow = A + (size_t)(row0 + c) * DM;

  v8f acc[4];
  #pragma unroll
  for (int t = 0; t < 4; ++t) acc[t] = (v8f){0,0,0,0,0,0,0,0};

  for (int ks = 0; ks < DM / 32; ++ks) {
    v16bf a = loadA(Arow, ks * 32 + hi * 8);
    #pragma unroll
    for (int t = 0; t < 4; ++t) {
      v16bf b = load16(Bt + (size_t)(colbase + t * 16 + c) * DM + ks * 32 + hi * 16);
      acc[t] = wmma_bf(a, b, acc[t]);
    }
  }

  const int n = blockIdx.x;
  if (mode == 0) {
    float s0[8], s1[8];
    #pragma unroll
    for (int v = 0; v < 8; ++v) { s0[v] = 0.f; s1[v] = 0.f; }
    #pragma unroll
    for (int t = 0; t < 4; ++t) {
      C8 u; u.v = acc[t];
      #pragma unroll
      for (int v = 0; v < 8; ++v) {
        int d = colbase + t * 16 + c;
        int m = row0 + v + 8 * hi, i0 = m >> 1, bb = m & 1;
        size_t o = (((size_t)bb * NH + n) * QLEN + i0) * DH + (t * 16 + c);
        float qv = u.f[v];
        o0[o] = f2bf(qv + b0[d]);
        o1[o] = f2bf(qv + b1[d]);
        float qs = qv + b2[d];
        s0[v] += qs * se[d];
        s1[v] += qs * se[NH * DH + d];
      }
    }
    #pragma unroll
    for (int v = 0; v < 8; ++v) {
      for (int o = 1; o < 16; o <<= 1) { s0[v] += __shfl_xor(s0[v], o); s1[v] += __shfl_xor(s1[v], o); }
      if (c == 0) {
        int m = row0 + v + 8 * hi, i0 = m >> 1, bb = m & 1;
        size_t e = (((size_t)bb * NH + n) * QLEN + i0) * 2;
        ef[e] = s0[v]; ef[e + 1] = s1[v];
      }
    }
  } else if (mode == 4) {
    #pragma unroll
    for (int t = 0; t < 4; ++t) {
      C8 u; u.v = acc[t];
      #pragma unroll
      for (int v = 0; v < 8; ++v) {
        int m = row0 + v + 8 * hi, col = colbase + t * 16 + c;
        resout[(size_t)m * DM + col] = u.f[v] + hres[(size_t)m * DM + col];
      }
    }
  } else {
    #pragma unroll
    for (int t = 0; t < 4; ++t) {
      C8 u; u.v = acc[t];
      #pragma unroll
      for (int v = 0; v < 8; ++v) {
        int m = row0 + v + 8 * hi, bb = m & 1, i0 = m >> 1, dl = t * 16 + c;
        size_t o;
        if (mode == 1)      o = (((size_t)bb * NH + n) * QLEN + i0) * DH + dl;           // K: [b][n][j][d]
        else if (mode == 2) o = (((size_t)bb * NH + n) * DH + dl) * QLEN + i0;           // V: [b][n][d][j]
        else                o = (((size_t)bb * NH + n) * RLEN + i0) * DH + dl;           // R: [b][n][jr][d]
        o0[o] = f2bf(u.f[v]);
      }
    }
  }
}

// ---------------- fused flash attention with relative shift ----------------
// block = (i-tile of 64 queries) x (b*NH+n); 4 waves, each owns 16 query rows.
__global__ __launch_bounds__(128) void attn_kernel(
    const unsigned short* __restrict__ qw, const unsigned short* __restrict__ qr,
    const unsigned short* __restrict__ kb, const unsigned short* __restrict__ vT,
    const unsigned short* __restrict__ kr, const float* __restrict__ ef,
    const unsigned* __restrict__ bits, unsigned short* __restrict__ av)
{
  extern __shared__ unsigned short smem[];
  unsigned short* strip = smem;                 // [64][1024] bf16: bd_raw strip (128 KB)
  unsigned short* Pl    = smem + 64 * 1024;     // [4][16][64] bf16: P staging (8 KB)

  const int w = threadIdx.x >> 5, lane = threadIdx.x & 31;
  const int c = lane & 15, hi = lane >> 4;
  const int it = blockIdx.x * 64;
  const int bn = blockIdx.y;                    // b*NH + n
  const int bb = bn >> 4, nh = bn & 15;

  const unsigned short* qwrow = qw + ((size_t)bn * QLEN + it + w * 16 + c) * DH;
  const unsigned short* qrrow = qr + ((size_t)bn * QLEN + it + w * 16 + c) * DH;
  v16bf Aqw[2], Aqr[2];
  #pragma unroll
  for (int ks = 0; ks < 2; ++ks) {
    Aqw[ks] = loadA(qwrow, ks * 32 + hi * 8);
    Aqr[ks] = loadA(qrrow, ks * 32 + hi * 8);
  }

  // bd strip: strip[di][x] = (q+rr_bias)[it+di] . kr[jrlo + x], jrlo = Q - it - 63.
  // Causality means only jr <= Q is ever read -> strip width it+64 <= 1024.
  const int nx = it / 64 + 1;
  const int jrlo = QLEN - it - 63;
  for (int xt = 0; xt < nx; ++xt) {
    C8 bd[4];
    #pragma unroll
    for (int t = 0; t < 4; ++t) bd[t].v = (v8f){0,0,0,0,0,0,0,0};
    #pragma unroll
    for (int t = 0; t < 4; ++t)
      #pragma unroll
      for (int ks = 0; ks < 2; ++ks) {
        const unsigned short* bp = kr + ((size_t)bn * RLEN + (jrlo + xt * 64 + t * 16 + c)) * DH
                                      + ks * 32 + hi * 16;
        bd[t].v = wmma_bf(Aqr[ks], load16(bp), bd[t].v);
      }
    #pragma unroll
    for (int t = 0; t < 4; ++t)
      #pragma unroll
      for (int v = 0; v < 8; ++v)
        strip[(w * 16 + v + 8 * hi) * 1024 + xt * 64 + t * 16 + c] = f2bf(bd[t].f[v]);
  }
  __syncthreads();

  float ef0v[8], ef1v[8];
  #pragma unroll
  for (int v = 0; v < 8; ++v) {
    int i0 = it + w * 16 + v + 8 * hi;
    ef0v[v] = ef[((size_t)bn * QLEN + i0) * 2];
    ef1v[v] = ef[((size_t)bn * QLEN + i0) * 2 + 1];
  }

  float mrow[8], lrow[8];
  C8 O[4];
  #pragma unroll
  for (int v = 0; v < 8; ++v) { mrow[v] = -3.0e38f; lrow[v] = 0.f; }
  #pragma unroll
  for (int t = 0; t < 4; ++t) O[t].v = (v8f){0,0,0,0,0,0,0,0};

  for (int jt = 0; jt <= it; jt += 64) {
    // ac = Qw . K^T
    C8 S[4];
    #pragma unroll
    for (int t = 0; t < 4; ++t) S[t].v = (v8f){0,0,0,0,0,0,0,0};
    #pragma unroll
    for (int t = 0; t < 4; ++t)
      #pragma unroll
      for (int ks = 0; ks < 2; ++ks) {
        const unsigned short* bp = kb + ((size_t)bn * QLEN + jt + t * 16 + c) * DH + ks * 32 + hi * 16;
        S[t].v = wmma_bf(Aqw[ks], load16(bp), S[t].v);
      }
    // + bd (shifted read) + segment term, scale, causal mask
    #pragma unroll
    for (int t = 0; t < 4; ++t) {
      int j = jt + t * 16 + c;
      #pragma unroll
      for (int v = 0; v < 8; ++v) {
        int di = w * 16 + v + 8 * hi, irow = it + di;
        int idx = j - irow + it + 63; if (idx > 1023) idx = 1023;   // masked lanes only
        float bdv = bf2f(strip[di * 1024 + idx]);
        unsigned word = bits[((size_t)bb * QLEN + irow) * 32 + (j >> 5)];
        float efv = ((word >> (j & 31)) & 1u) ? ef1v[v] : ef0v[v];
        float s = (S[t].f[v] + bdv + efv) * 0.125f;
        if (j > irow) s -= 1.0e30f;
        S[t].f[v] = s;
      }
    }
    // online softmax (rows live on one 16-lane half)
    #pragma unroll
    for (int v = 0; v < 8; ++v) {
      float rmax = S[0].f[v];
      #pragma unroll
      for (int t = 1; t < 4; ++t) rmax = fmaxf(rmax, S[t].f[v]);
      for (int o = 1; o < 16; o <<= 1) rmax = fmaxf(rmax, __shfl_xor(rmax, o));
      float mnew = fmaxf(mrow[v], rmax);
      float alpha = __expf(mrow[v] - mnew);
      float ps = 0.f;
      #pragma unroll
      for (int t = 0; t < 4; ++t) { float p = __expf(S[t].f[v] - mnew); S[t].f[v] = p; ps += p; }
      for (int o = 1; o < 16; o <<= 1) ps += __shfl_xor(ps, o);
      lrow[v] = lrow[v] * alpha + ps;
      mrow[v] = mnew;
      #pragma unroll
      for (int td = 0; td < 4; ++td) O[td].f[v] *= alpha;
    }
    // stage P to LDS (C-layout -> A-layout), then O += P . V
    #pragma unroll
    for (int t = 0; t < 4; ++t)
      #pragma unroll
      for (int v = 0; v < 8; ++v)
        Pl[(w * 16 + v + 8 * hi) * 64 + t * 16 + c] = f2bf(S[t].f[v]);
    __syncthreads();
    v16bf Ap[2];
    #pragma unroll
    for (int ks = 0; ks < 2; ++ks) Ap[ks] = loadA(Pl + (w * 16 + c) * 64, ks * 32 + hi * 8);
    #pragma unroll
    for (int td = 0; td < 4; ++td)
      #pragma unroll
      for (int ks = 0; ks < 2; ++ks) {
        const unsigned short* bp = vT + ((size_t)bn * DH + td * 16 + c) * QLEN + jt + ks * 32 + hi * 16;
        O[td].v = wmma_bf(Ap[ks], load16(bp), O[td].v);
      }
    __syncthreads();
  }

  // normalize and write attn_vec as bf16 rows [m = i*B+b][n*64+d]
  #pragma unroll
  for (int td = 0; td < 4; ++td)
    #pragma unroll
    for (int v = 0; v < 8; ++v) {
      int irow = it + w * 16 + v + 8 * hi;
      int m = irow * BSZ + bb;
      av[(size_t)m * DM + nh * DH + td * 16 + c] = f2bf(O[td].f[v] / lrow[v]);
    }
}

// ---------------- LayerNorm ----------------
__global__ __launch_bounds__(256) void ln_kernel(const float* __restrict__ res,
                                                 const float* __restrict__ g,
                                                 const float* __restrict__ be,
                                                 float* __restrict__ out) {
  __shared__ float red[8];
  int m = blockIdx.x;
  const float* row = res + (size_t)m * DM;
  float x[4]; float s = 0.f;
  #pragma unroll
  for (int k = 0; k < 4; ++k) { x[k] = row[threadIdx.x + 256 * k]; s += x[k]; }
  for (int o = 1; o < 32; o <<= 1) s += __shfl_xor(s, o);
  if ((threadIdx.x & 31) == 0) red[threadIdx.x >> 5] = s;
  __syncthreads();
  float tot = 0.f;
  #pragma unroll
  for (int r = 0; r < 8; ++r) tot += red[r];
  float mu = tot * (1.0f / DM);
  float vs = 0.f;
  #pragma unroll
  for (int k = 0; k < 4; ++k) { float d = x[k] - mu; vs += d * d; }
  for (int o = 1; o < 32; o <<= 1) vs += __shfl_xor(vs, o);
  __syncthreads();
  if ((threadIdx.x & 31) == 0) red[threadIdx.x >> 5] = vs;
  __syncthreads();
  float v2 = 0.f;
  #pragma unroll
  for (int r = 0; r < 8; ++r) v2 += red[r];
  float inv = rsqrtf(v2 * (1.0f / DM) + 1e-12f);
  #pragma unroll
  for (int k = 0; k < 4; ++k) {
    int idx = threadIdx.x + 256 * k;
    out[(size_t)m * DM + idx] = (x[k] - mu) * inv * g[idx] + be[idx];
  }
}

// ---------------- workspace layout (bytes) ----------------
static const size_t OFF_HBF = 0;
static const size_t OFF_RBF = OFF_HBF + 2048ull * 1024 * 2;
static const size_t OFF_WTQ = OFF_RBF + 4096ull * 1024 * 2;
static const size_t OFF_WTK = OFF_WTQ + 1024ull * 1024 * 2;
static const size_t OFF_WTV = OFF_WTK + 1024ull * 1024 * 2;
static const size_t OFF_WTR = OFF_WTV + 1024ull * 1024 * 2;
static const size_t OFF_WOB = OFF_WTR + 1024ull * 1024 * 2;
static const size_t OFF_QW  = OFF_WOB + 1024ull * 1024 * 2;
static const size_t OFF_QR  = OFF_QW + 2ull * NH * QLEN * DH * 2;
static const size_t OFF_KB  = OFF_QR + 2ull * NH * QLEN * DH * 2;
static const size_t OFF_VT  = OFF_KB + 2ull * NH * QLEN * DH * 2;
static const size_t OFF_KR  = OFF_VT + 2ull * NH * QLEN * DH * 2;
static const size_t OFF_EF  = OFF_KR + 2ull * NH * RLEN * DH * 2;
static const size_t OFF_BIT = OFF_EF + 2ull * NH * QLEN * 2 * 4;
static const size_t OFF_AV  = OFF_BIT + 2ull * QLEN * 32 * 4;
static const size_t OFF_RES = OFF_AV + 2048ull * 1024 * 2;

extern "C" void kernel_launch(void* const* d_in, const int* in_sizes, int n_in,
                              void* d_out, int out_size, void* d_ws, size_t ws_size,
                              hipStream_t stream) {
  (void)in_sizes; (void)n_in; (void)out_size; (void)ws_size;
  const float* h   = (const float*)d_in[0];
  const float* r   = (const float*)d_in[1];
  const float* seg = (const float*)d_in[2];
  const float* Wq  = (const float*)d_in[4];
  const float* Wk  = (const float*)d_in[5];
  const float* Wv  = (const float*)d_in[6];
  const float* Wr  = (const float*)d_in[7];
  const float* Wo  = (const float*)d_in[8];
  const float* rwb = (const float*)d_in[9];
  const float* rrb = (const float*)d_in[10];
  const float* rsb = (const float*)d_in[11];
  const float* se  = (const float*)d_in[12];
  const float* gam = (const float*)d_in[13];
  const float* bet = (const float*)d_in[14];
  float* out = (float*)d_out;

  char* ws = (char*)d_ws;
  unsigned short* h_bf = (unsigned short*)(ws + OFF_HBF);
  unsigned short* r_bf = (unsigned short*)(ws + OFF_RBF);
  unsigned short* wtq  = (unsigned short*)(ws + OFF_WTQ);
  unsigned short* wtk  = (unsigned short*)(ws + OFF_WTK);
  unsigned short* wtv  = (unsigned short*)(ws + OFF_WTV);
  unsigned short* wtr  = (unsigned short*)(ws + OFF_WTR);
  unsigned short* wob  = (unsigned short*)(ws + OFF_WOB);
  unsigned short* qw   = (unsigned short*)(ws + OFF_QW);
  unsigned short* qr   = (unsigned short*)(ws + OFF_QR);
  unsigned short* kb   = (unsigned short*)(ws + OFF_KB);
  unsigned short* vT   = (unsigned short*)(ws + OFF_VT);
  unsigned short* kr   = (unsigned short*)(ws + OFF_KR);
  float*          ef   = (float*)(ws + OFF_EF);
  unsigned*       bits = (unsigned*)(ws + OFF_BIT);
  unsigned short* av   = (unsigned short*)(ws + OFF_AV);
  float*          res  = (float*)(ws + OFF_RES);

  // prep: bf16 conversions, weight transposes, segment bitset
  convert_bf16<<<(2048 * 1024) / 256, 256, 0, stream>>>(h, h_bf, 2048 * 1024);
  convert_bf16<<<(4096 * 1024) / 256, 256, 0, stream>>>(r, r_bf, 4096 * 1024);
  convert_bf16<<<(1024 * 1024) / 256, 256, 0, stream>>>(Wo, wob, 1024 * 1024);
  transpose_conv<<<dim3(32, 32), 256, 0, stream>>>(Wq, wtq);
  transpose_conv<<<dim3(32, 32), 256, 0, stream>>>(Wk, wtk);
  transpose_conv<<<dim3(32, 32), 256, 0, stream>>>(Wv, wtv);
  transpose_conv<<<dim3(32, 32), 256, 0, stream>>>(Wr, wtr);
  segbits_kernel<<<(BSZ * QLEN * 32) / 256, 256, 0, stream>>>(seg, bits);

  // head projections (WMMA)
  wmma_gemm_kernel<<<dim3(NH, 32), 128, 0, stream>>>(h_bf, wtq, 0, qw, qr, ef, rwb, rrb, rsb, se, nullptr, nullptr);
  wmma_gemm_kernel<<<dim3(NH, 32), 128, 0, stream>>>(h_bf, wtk, 1, kb, nullptr, nullptr, nullptr, nullptr, nullptr, nullptr, nullptr, nullptr);
  wmma_gemm_kernel<<<dim3(NH, 32), 128, 0, stream>>>(h_bf, wtv, 2, vT, nullptr, nullptr, nullptr, nullptr, nullptr, nullptr, nullptr, nullptr);
  wmma_gemm_kernel<<<dim3(NH, 64), 128, 0, stream>>>(r_bf, wtr, 3, kr, nullptr, nullptr, nullptr, nullptr, nullptr, nullptr, nullptr, nullptr);

  // fused relative-attention (WMMA flash) — 128 KB bd strip + 8 KB P staging in LDS
  size_t smem = (64ull * 1024 + 4 * 16 * 64) * sizeof(unsigned short);
  attn_kernel<<<dim3(QLEN / 64, BSZ * NH), 128, smem, stream>>>(qw, qr, kb, vT, kr, ef, bits, av);

  // output projection + residual (WMMA), then LayerNorm
  wmma_gemm_kernel<<<dim3(DM / 64, 32), 128, 0, stream>>>(av, wob, 4, nullptr, nullptr, nullptr, nullptr, nullptr, nullptr, nullptr, h, res);
  ln_kernel<<<2048, 256, 0, stream>>>(res, gam, bet, out);
}
